// HierDSFeedForward_83803401879936
// MI455X (gfx1250) — matrix operand order repaired and checked by reading, not compiled, not run on hardware
//
#include <hip/hip_runtime.h>
#include <hip/hip_bf16.h>

// Problem constants (match reference)
#define S_TOK 8192
#define C_DIM 512
#define H_DIM 1024
#define G_NUM 2
#define EG_NUM 4
#define E_NUM 8
#define LN_EPSF 1e-5f

typedef __attribute__((ext_vector_type(16))) _Float16 v16h;
typedef __attribute__((ext_vector_type(8)))  _Float16 v8h;
typedef __attribute__((ext_vector_type(8)))  float    v8f;

__device__ __forceinline__ v8f vzero8() {
    v8f z;
#pragma unroll
    for (int i = 0; i < 8; ++i) z[i] = 0.f;
    return z;
}

// A-matrix fragment (16x32 f16, row-major source, this lane's row base + kk
// already applied). ISA layout: lane-half hi: VGPR0-3 = K[8*hi .. 8*hi+7],
// VGPR4-7 = K[16+8*hi .. 16+8*hi+7].
__device__ __forceinline__ v16h load_fragA(const _Float16* rowk, int hi) {
    const _Float16* q = rowk + 8 * hi;
    v8h c0 = *(const v8h*)(q);
    v8h c1 = *(const v8h*)(q + 16);
    v16h r;
#pragma unroll
    for (int i = 0; i < 8; ++i) { r[i] = c0[i]; r[i + 8] = c1[i]; }
    return r;
}

// B-matrix fragment (32x16 f16). Source is W[n, k] row-major (so B[k][n] =
// W[n][k]); lane owns column n, lane-half hi holds K[8*hi..8*hi+7], K[16+8*hi..16+8*hi+7].
__device__ __forceinline__ v16h load_fragB(const _Float16* rowk, int hi) {
    const _Float16* q = rowk + 8 * hi;
    v8h c0 = *(const v8h*)(q);
    v8h c1 = *(const v8h*)(q + 16);
    v16h r;
#pragma unroll
    for (int i = 0; i < 8; ++i) { r[i] = c0[i]; r[i + 8] = c1[i]; }
    return r;
}

__device__ __forceinline__ v8f wmma16(v16h a, v16h b, v8f c) {
    return __builtin_amdgcn_wmma_f32_16x16x32_f16(false, a, false, b,
                                                  (short)0, c, false, false);
}

// ---------------------------------------------------------------- utilities
__global__ void moe_zero_counts(int* counts) {
    if (threadIdx.x < E_NUM) counts[threadIdx.x] = 0;
}

__global__ void moe_cvt_f16(const float* __restrict__ src,
                            _Float16* __restrict__ dst, int n) {
    int i = blockIdx.x * blockDim.x + threadIdx.x;
    if (i < n) dst[i] = (_Float16)src[i];
}

// ------------------------------------------------------------- LayerNorm
// one wave (32 lanes) per token; f16 output
__global__ void moe_layernorm(const float* __restrict__ x,
                              const float* __restrict__ scale,
                              const float* __restrict__ bias,
                              _Float16* __restrict__ flatH) {
    int gtid = blockIdx.x * blockDim.x + threadIdx.x;
    int tok  = gtid >> 5;
    int lane = threadIdx.x & 31;
    if (tok >= S_TOK) return;
    const float* xr = x + (size_t)tok * C_DIM;
    float s1 = 0.f, s2 = 0.f;
    for (int k = lane; k < C_DIM; k += 32) {
        float v = xr[k];
        s1 += v; s2 += v * v;
    }
#pragma unroll
    for (int off = 16; off > 0; off >>= 1) {
        s1 += __shfl_xor(s1, off, 32);
        s2 += __shfl_xor(s2, off, 32);
    }
    float mu  = s1 * (1.f / C_DIM);
    float var = s2 * (1.f / C_DIM) - mu * mu;
    float rs  = rsqrtf(var + LN_EPSF);
    for (int k = lane; k < C_DIM; k += 32) {
        float v = (xr[k] - mu) * rs * scale[k] + bias[k];
        flatH[(size_t)tok * C_DIM + k] = (_Float16)v;
    }
}

// ------------------------------------------------------------- Gating
// one thread per token: group argmax, softmax over EG, top-2, renorm,
// build per-expert gather lists.
__global__ void moe_gating(const _Float16* __restrict__ flatH,
                           const float* __restrict__ ggw,   // [G, C]
                           const float* __restrict__ egw,   // [E, C]
                           const float* __restrict__ gbias, // [G]
                           const float* __restrict__ ebias, // [E]
                           int* __restrict__ counts,        // [E]
                           int* __restrict__ tokList,       // [E*S]
                           int* __restrict__ slotList,      // [E*S]
                           float* __restrict__ wList) {     // [E*S]
    int s = blockIdx.x * blockDim.x + threadIdx.x;
    if (s >= S_TOK) return;
    const _Float16* xr = flatH + (size_t)s * C_DIM;
    float gl0 = 0.f, gl1 = 0.f;
    float el[E_NUM];
#pragma unroll
    for (int e = 0; e < E_NUM; ++e) el[e] = 0.f;
    for (int k = 0; k < C_DIM; ++k) {
        float xv = (float)xr[k];
        gl0 += xv * ggw[k];
        gl1 += xv * ggw[C_DIM + k];
#pragma unroll
        for (int e = 0; e < E_NUM; ++e) el[e] += xv * egw[e * C_DIM + k];
    }
    gl0 += gbias[0]; gl1 += gbias[1];
    int g = (gl1 > gl0) ? 1 : 0;               // argmax, first max wins
    float lo[EG_NUM];
    float m = -3.4e38f;
#pragma unroll
    for (int j = 0; j < EG_NUM; ++j) {
        lo[j] = el[g * EG_NUM + j] + ebias[g * EG_NUM + j];
        m = fmaxf(m, lo[j]);
    }
    float p[EG_NUM]; float sum = 0.f;
#pragma unroll
    for (int j = 0; j < EG_NUM; ++j) { p[j] = expf(lo[j] - m); sum += p[j]; }
    float inv = 1.f / sum;
#pragma unroll
    for (int j = 0; j < EG_NUM; ++j) p[j] *= inv;
    // top-2 (first max wins on ties, like jax top_k)
    int i1 = 0; float v1 = p[0];
#pragma unroll
    for (int j = 1; j < EG_NUM; ++j) if (p[j] > v1) { v1 = p[j]; i1 = j; }
    int i2 = -1; float v2 = -3.4e38f;
#pragma unroll
    for (int j = 0; j < EG_NUM; ++j)
        if (j != i1 && p[j] > v2) { v2 = p[j]; i2 = j; }
    float norm = 1.f / (v1 + v2 + 1e-8f);
    float w1 = v1 * norm, w2 = v2 * norm;
    int e1 = g * EG_NUM + i1;
    int e2 = g * EG_NUM + i2;
    int s1i = atomicAdd(&counts[e1], 1);
    tokList[e1 * S_TOK + s1i] = s;
    slotList[e1 * S_TOK + s1i] = 0;
    wList[e1 * S_TOK + s1i] = w1;
    int s2i = atomicAdd(&counts[e2], 1);
    tokList[e2 * S_TOK + s2i] = s;
    slotList[e2 * S_TOK + s2i] = 1;
    wList[e2 * S_TOK + s2i] = w2;
}

// ---------------------------------------------- in-projection + SwiGLU GEMM
// A: flatH [S, C] f16; W: [2H, C] f16; Hout: [S, H] f16.
// Per-wave tile 32M x 32N(h): 2 M-subtiles x 2 N-subtiles x (a,b) halves
// = 8 accumulators (64 VGPRs, no spills); each B fragment feeds 2 WMMAs.
__global__ void __launch_bounds__(256, 1)
moe_inproj_swiglu(const _Float16* __restrict__ A,
                  const _Float16* __restrict__ W,
                  _Float16* __restrict__ Hout) {
    int wave = threadIdx.x >> 5;
    int lane = threadIdx.x & 31;
    int hi = lane >> 4;
    int ln = lane & 15;
    int m0 = (blockIdx.x * 8 + wave) * 32;
    int n0 = blockIdx.y * 32;
    v8f accA[2][2], accB[2][2];
#pragma unroll
    for (int u = 0; u < 2; ++u)
#pragma unroll
        for (int t = 0; t < 2; ++t) { accA[u][t] = vzero8(); accB[u][t] = vzero8(); }
    const _Float16* arow0 = A + (size_t)(m0 + ln) * C_DIM;
    const _Float16* arow1 = A + (size_t)(m0 + 16 + ln) * C_DIM;
    for (int kk = 0; kk < C_DIM; kk += 32) {
        v16h af0 = load_fragA(arow0 + kk, hi);
        v16h af1 = load_fragA(arow1 + kk, hi);
#pragma unroll
        for (int t = 0; t < 2; ++t) {
            int col = n0 + t * 16 + ln;
            v16h ba = load_fragB(W + (size_t)col * C_DIM + kk, hi);
            accA[0][t] = wmma16(af0, ba, accA[0][t]);
            accA[1][t] = wmma16(af1, ba, accA[1][t]);
            v16h bb = load_fragB(W + (size_t)(col + H_DIM) * C_DIM + kk, hi);
            accB[0][t] = wmma16(af0, bb, accB[0][t]);
            accB[1][t] = wmma16(af1, bb, accB[1][t]);
        }
    }
#pragma unroll
    for (int u = 0; u < 2; ++u) {
#pragma unroll
        for (int t = 0; t < 2; ++t) {
            int col = n0 + t * 16 + ln;
#pragma unroll
            for (int v = 0; v < 8; ++v) {
                int row = m0 + u * 16 + v + 8 * hi;
                float a = accA[u][t][v], b = accB[u][t][v];
                float h = (a / (1.f + __expf(-a))) * b;   // silu(a)*b
                Hout[(size_t)row * H_DIM + col] = (_Float16)h;
            }
        }
    }
}

// ---------------------------------------------- shared out-projection GEMM
// out[s,c] = hS[s,:] . Wso[c,:] + b[c]   (A:[S,H], W:[C,H], out f32 [S,C])
// Per-wave tile 32M x 64N: 8 accumulators; B fragments feed 2 WMMAs each.
__global__ void __launch_bounds__(256, 1)
moe_shared_out(const _Float16* __restrict__ A,
               const _Float16* __restrict__ W,
               const float* __restrict__ bias,
               float* __restrict__ out) {
    int wave = threadIdx.x >> 5;
    int lane = threadIdx.x & 31;
    int hi = lane >> 4;
    int ln = lane & 15;
    int m0 = (blockIdx.x * 8 + wave) * 32;
    int n0 = blockIdx.y * 64;
    v8f acc[2][4];
#pragma unroll
    for (int u = 0; u < 2; ++u)
#pragma unroll
        for (int t = 0; t < 4; ++t) acc[u][t] = vzero8();
    const _Float16* arow0 = A + (size_t)(m0 + ln) * H_DIM;
    const _Float16* arow1 = A + (size_t)(m0 + 16 + ln) * H_DIM;
    for (int kk = 0; kk < H_DIM; kk += 32) {
        v16h af0 = load_fragA(arow0 + kk, hi);
        v16h af1 = load_fragA(arow1 + kk, hi);
#pragma unroll
        for (int t = 0; t < 4; ++t) {
            int col = n0 + t * 16 + ln;
            v16h bf = load_fragB(W + (size_t)col * H_DIM + kk, hi);
            acc[0][t] = wmma16(af0, bf, acc[0][t]);
            acc[1][t] = wmma16(af1, bf, acc[1][t]);
        }
    }
#pragma unroll
    for (int u = 0; u < 2; ++u) {
#pragma unroll
        for (int t = 0; t < 4; ++t) {
            int col = n0 + t * 16 + ln;
            float b = bias[col];
#pragma unroll
            for (int v = 0; v < 8; ++v) {
                int row = m0 + u * 16 + v + 8 * hi;
                out[(size_t)row * C_DIM + col] = acc[u][t][v] + b;
            }
        }
    }
}

// ---------------------------------------------- gathered expert GEMM
// For expert e: rows = gathered tokens; contribution w*(hE@W_e^T)+bias_e is
// written to contrib[(tok*2 + slot)*C + col]  (no atomics -> deterministic).
__global__ void __launch_bounds__(256, 1)
moe_expert_out(const _Float16* __restrict__ hE,
               const _Float16* __restrict__ Wexp, // [E,C,H]
               const float* __restrict__ ebias,   // [E,C]
               const int* __restrict__ counts,
               const int* __restrict__ tokList,
               const int* __restrict__ slotList,
               const float* __restrict__ wList,
               float* __restrict__ contrib) {     // [S,2,C]
    int e = blockIdx.z;
    int cnt = counts[e];
    int wave = threadIdx.x >> 5;
    int lane = threadIdx.x & 31;
    int hi = lane >> 4;
    int ln = lane & 15;
    int m0 = (blockIdx.x * 8 + wave) * 32;
    if (m0 >= cnt) return;            // wave-uniform exit, EXEC stays full
    int n0 = blockIdx.y * 64;

    int tok[2], slot[2], ok[2];
    float wgt[2];
#pragma unroll
    for (int u = 0; u < 2; ++u) {
        int rIdx = m0 + u * 16 + ln;
        ok[u] = (rIdx < cnt) ? 1 : 0;
        int base = e * S_TOK + (ok[u] ? rIdx : 0);
        tok[u]  = ok[u] ? tokList[base]  : 0;
        slot[u] = ok[u] ? slotList[base] : 0;
        wgt[u]  = ok[u] ? wList[base]    : 0.f;
    }

    const _Float16* arow0 = hE + (size_t)tok[0] * H_DIM;
    const _Float16* arow1 = hE + (size_t)tok[1] * H_DIM;
    const _Float16* Wb = Wexp + (size_t)e * C_DIM * H_DIM;
    v8f acc[2][4];
#pragma unroll
    for (int u = 0; u < 2; ++u)
#pragma unroll
        for (int t = 0; t < 4; ++t) acc[u][t] = vzero8();
    for (int kk = 0; kk < H_DIM; kk += 32) {
        v16h af0 = load_fragA(arow0 + kk, hi);
        v16h af1 = load_fragA(arow1 + kk, hi);
#pragma unroll
        for (int t = 0; t < 4; ++t) {
            int col = n0 + t * 16 + ln;
            v16h bf = load_fragB(Wb + (size_t)col * H_DIM + kk, hi);
            acc[0][t] = wmma16(af0, bf, acc[0][t]);
            acc[1][t] = wmma16(af1, bf, acc[1][t]);
        }
    }
#pragma unroll
    for (int u = 0; u < 2; ++u) {
#pragma unroll
        for (int t = 0; t < 4; ++t) {
            int col = n0 + t * 16 + ln;
            float bcol = ebias[e * C_DIM + col];
#pragma unroll
            for (int v = 0; v < 8; ++v) {
                int srcl = v + 8 * hi;             // lane holding row v+8*hi
                int   tv  = __shfl(tok[u],  srcl, 32);
                int   sv  = __shfl(slot[u], srcl, 32);
                float wv  = __shfl(wgt[u],  srcl, 32);
                int   okv = __shfl(ok[u],   srcl, 32);
                if (okv) {
                    contrib[((size_t)tv * 2 + sv) * C_DIM + col] =
                        acc[u][t][v] * wv + bcol;
                }
            }
        }
    }
}

// ---------------------------------------------- final combine
__global__ void moe_combine(float* __restrict__ out,
                            const float* __restrict__ contrib) {
    int i = blockIdx.x * blockDim.x + threadIdx.x;
    if (i >= S_TOK * C_DIM) return;
    int s = i / C_DIM;
    int c = i - s * C_DIM;
    size_t b = (size_t)s * 2 * C_DIM + c;
    out[i] = out[i] + contrib[b] + contrib[b + C_DIM];
}

// ================================================================ launcher
extern "C" void kernel_launch(void* const* d_in, const int* in_sizes, int n_in,
                              void* d_out, int out_size, void* d_ws, size_t ws_size,
                              hipStream_t stream) {
    const float* x        = (const float*)d_in[0];
    const float* ln_scale = (const float*)d_in[1];
    const float* ln_bias  = (const float*)d_in[2];
    const float* w_si     = (const float*)d_in[3];   // [2H, C]
    const float* w_so     = (const float*)d_in[4];   // [C, H]
    const float* b_so     = (const float*)d_in[5];   // [C]
    const float* w_ei     = (const float*)d_in[6];   // [2H, C]
    const float* w_eo     = (const float*)d_in[7];   // [E, C, H]
    const float* b_eo     = (const float*)d_in[8];   // [E, C]
    const float* ggw      = (const float*)d_in[9];   // [G, C]
    const float* egw      = (const float*)d_in[10];  // [E, C]
    const float* gbias    = (const float*)d_in[11];  // [G]
    const float* ebias    = (const float*)d_in[12];  // [E]
    float* out = (float*)d_out;

    // workspace carve-up
    char* p = (char*)d_ws;
    _Float16* flatH = (_Float16*)p; p += (size_t)S_TOK * C_DIM * 2;       // 8 MB
    _Float16* hS    = (_Float16*)p; p += (size_t)S_TOK * H_DIM * 2;       // 16 MB
    _Float16* hE    = (_Float16*)p; p += (size_t)S_TOK * H_DIM * 2;       // 16 MB
    _Float16* wSI   = (_Float16*)p; p += (size_t)2 * H_DIM * C_DIM * 2;   // 2 MB
    _Float16* wEI   = (_Float16*)p; p += (size_t)2 * H_DIM * C_DIM * 2;   // 2 MB
    _Float16* wSO   = (_Float16*)p; p += (size_t)C_DIM * H_DIM * 2;       // 1 MB
    _Float16* wEO   = (_Float16*)p; p += (size_t)E_NUM * C_DIM * H_DIM * 2; // 8 MB
    float* contrib  = (float*)p;    p += (size_t)S_TOK * 2 * C_DIM * 4;   // 32 MB
    int* counts     = (int*)p;      p += 256;
    int* tokList    = (int*)p;      p += (size_t)E_NUM * S_TOK * 4;
    int* slotList   = (int*)p;      p += (size_t)E_NUM * S_TOK * 4;
    float* wList    = (float*)p;    p += (size_t)E_NUM * S_TOK * 4;

    const int TPB = 256;
    // 0) zero expert counts
    moe_zero_counts<<<1, 32, 0, stream>>>(counts);
    // 1) weight converts f32 -> f16
    {
        int n1 = 2 * H_DIM * C_DIM;
        moe_cvt_f16<<<(n1 + TPB - 1) / TPB, TPB, 0, stream>>>(w_si, wSI, n1);
        moe_cvt_f16<<<(n1 + TPB - 1) / TPB, TPB, 0, stream>>>(w_ei, wEI, n1);
        int n2 = C_DIM * H_DIM;
        moe_cvt_f16<<<(n2 + TPB - 1) / TPB, TPB, 0, stream>>>(w_so, wSO, n2);
        int n3 = E_NUM * C_DIM * H_DIM;
        moe_cvt_f16<<<(n3 + TPB - 1) / TPB, TPB, 0, stream>>>(w_eo, wEO, n3);
    }
    // 2) LayerNorm -> f16 activations (one wave per token)
    moe_layernorm<<<S_TOK / 8, TPB, 0, stream>>>(x, ln_scale, ln_bias, flatH);
    // 3) gating + gather-list build
    moe_gating<<<S_TOK / TPB, TPB, 0, stream>>>(flatH, ggw, egw, gbias, ebias,
                                                counts, tokList, slotList, wList);
    // 4) in-projections + SwiGLU (shared and expert), WMMA f16
    {
        dim3 g(S_TOK / 256, H_DIM / 32);
        moe_inproj_swiglu<<<g, TPB, 0, stream>>>(flatH, wSI, hS);
        moe_inproj_swiglu<<<g, TPB, 0, stream>>>(flatH, wEI, hE);
    }
    // 5) shared out-projection -> d_out
    {
        dim3 g(S_TOK / 256, C_DIM / 64);
        moe_shared_out<<<g, TPB, 0, stream>>>(hS, wSO, b_so, out);
    }
    // 6) gathered expert out-projections -> contrib slots
    {
        dim3 g(S_TOK / 256, C_DIM / 64, E_NUM);
        moe_expert_out<<<g, TPB, 0, stream>>>(hE, wEO, b_eo, counts,
                                              tokList, slotList, wList, contrib);
    }
    // 7) combine
    {
        int n = S_TOK * C_DIM;
        moe_combine<<<(n + TPB - 1) / TPB, TPB, 0, stream>>>(out, contrib);
    }
}
